// GraphTargetMessage_81698867904548
// MI455X (gfx1250) — matrix operand order repaired and checked by reading, not compiled, not run on hardware
//
#include <hip/hip_runtime.h>
#include <hip/hip_bf16.h>

typedef __attribute__((ext_vector_type(16))) _Float16 v16h;
typedef __attribute__((ext_vector_type(8)))  float    v8f;

#define BSZ  128
#define TSEQ 200
#define NSK  512
#define EDIM 64
#define LAG  30
#define TP   (TSEQ - LAG)      // 170
#define RR   (BSZ * TP)        // 21760

// ---- workspace layout ----
// [0, CROSS_BYTES)            : cross_effect, RR x 64 f32
// [CROSS_BYTES, +P_TOTAL*2)   : packed f16 weights in WMMA B-fragment order
#define CROSS_BYTES ((size_t)RR * EDIM * sizeof(float))   // 5,570,560 (32B aligned)
#define P_IVW0 0        // 128x64 -> 8192 halves
#define P_IVW1 8192     //  64x64 -> 4096
#define P_IVWF 12288    //  64x64 -> 4096
#define P_FOW0 16384    //  64x64 -> 4096
#define P_FOW1 20480    //  64x64 -> 4096
#define P_FOWF 24576    //  64x16(pad of 64x2) -> 1024
#define P_TOTAL 25600

// -------------------------------------------------------------------------
// Pack fp32 weights (row-major KxN) into f16 WMMA B-fragment order:
// frag f=(kt*ntiles+nt): lane holds column n=nt*16+(lane&15); element e of the
// lane's v16h is K = kt*32 + (lane>>4)*8 + e for e<8, kt*32 + 16 + (lane>>4)*8 + (e-8) for e>=8.
// -------------------------------------------------------------------------
__global__ void pack_weights(const float* ivW0, const float* ivW1, const float* ivWf,
                             const float* foW0, const float* foW1, const float* foWf,
                             _Float16* dst) {
    int g = blockIdx.x * blockDim.x + threadIdx.x;
    if (g >= P_TOTAL) return;
    const float* W; int Nsrc, Npad, base;
    if      (g < P_IVW1) { W = ivW0; Nsrc = 64; Npad = 64; base = P_IVW0; }
    else if (g < P_IVWF) { W = ivW1; Nsrc = 64; Npad = 64; base = P_IVW1; }
    else if (g < P_FOW0) { W = ivWf; Nsrc = 64; Npad = 64; base = P_IVWF; }
    else if (g < P_FOW1) { W = foW0; Nsrc = 64; Npad = 64; base = P_FOW0; }
    else if (g < P_FOWF) { W = foW1; Nsrc = 64; Npad = 64; base = P_FOW1; }
    else                 { W = foWf; Nsrc = 2;  Npad = 16; base = P_FOWF; }
    int loc    = g - base;
    int f      = loc >> 9;          // 512 halves per fragment
    int lane   = (loc >> 4) & 31;
    int e      = loc & 15;
    int ntiles = Npad >> 4;
    int kt = f / ntiles, nt = f % ntiles;
    int n  = nt * 16 + (lane & 15);
    int k  = kt * 32 + (lane >> 4) * 8 + (e < 8 ? e : 16 + (e - 8));
    float v = (n < Nsrc) ? W[k * Nsrc + n] : 0.0f;
    dst[g] = (_Float16)v;
}

// A-fragment for V_WMMA_F32_16X16X32_F16 from a row-major f16 row in LDS.
// lane m = lane&15 holds row m; kg = lane>>4. Elements 0..7 = K kg*8..kg*8+7,
// elements 8..15 = K 16+kg*8 .. 16+kg*8+7 (per ISA 16-bit A layout).
__device__ inline v16h load_a_frag(const _Float16* xrow, int kt, int kg) {
    const _Float16* p = xrow + kt * 32 + kg * 8;
    v16h a;
#pragma unroll
    for (int e = 0; e < 8; ++e) { a[e] = p[e]; a[8 + e] = p[16 + e]; }
    return a;
}

__device__ inline float leaky(float x) { return x >= 0.0f ? x : 0.01f * x; }

// LayerNorm 32 rows x 64 cols (fp32 in H) -> f16 in Xn, with gamma/beta.
__device__ inline void layernorm_to_f16(const float* H, _Float16* Xn,
                                        const float* g, const float* b,
                                        float* sMean, float* sRstd, int tid) {
    if (tid < 32) {
        const float* row = H + tid * 64;
        float s = 0.0f;
        for (int c = 0; c < 64; ++c) s += row[c];
        float m = s * (1.0f / 64.0f);
        float v = 0.0f;
        for (int c = 0; c < 64; ++c) { float d = row[c] - m; v += d * d; }
        sMean[tid] = m;
        sRstd[tid] = rsqrtf(v * (1.0f / 64.0f) + 1e-5f);
    }
    __syncthreads();
    for (int idx = tid; idx < 32 * 64; idx += 256) {
        int rr = idx >> 6, c = idx & 63;
        float x = (H[idx] - sMean[rr]) * sRstd[rr] * g[c] + b[c];
        Xn[idx] = (_Float16)x;
    }
    __syncthreads();
}

// -------------------------------------------------------------------------
// Kernel 1: one workgroup per target (b,t). Build edge features, run the
// edge MLP with WMMA, weighted-sum over L -> cross_effect row (64 f32).
// 256 threads = 8 waves; wave w owns output tile (mt = w>>2, nt = w&3).
// -------------------------------------------------------------------------
__global__ __launch_bounds__(256)
void edge_msg_kernel(const int* __restrict__ skill, const int* __restrict__ time_s,
                     const int* __restrict__ label,
                     const float* __restrict__ adj, const float* __restrict__ emb,
                     const float* __restrict__ b0, const float* __restrict__ g1,
                     const float* __restrict__ be1, const float* __restrict__ b1,
                     const float* __restrict__ gf, const float* __restrict__ bef,
                     const float* __restrict__ bfv,
                     const _Float16* __restrict__ pk,
                     float* __restrict__ cross) {
    __shared__ _Float16 sX[32 * 128];   // edge features; reused as Xn[32*64]
    __shared__ float    sH[32 * 64];    // H1 / H2 / Mout
    __shared__ float    sW[32];
    __shared__ float    sMean[32], sRstd[32];

    const int r = blockIdx.x;
    const int b = r / TP, t = r % TP;
    const int tid  = threadIdx.x;
    const int wave = tid >> 5, lane = tid & 31;
    const int mt = wave >> 2, nt = wave & 3;
    const int laneM = lane & 15, kg = lane >> 4;

    const int   st  = skill[b * TSEQ + t + LAG];
    const float ttg = (float)time_s[b * TSEQ + t + LAG];

    // ---- build X: rows l<30 = [emb[st] | emb[sh_l]*lab_l], rows 30,31 = 0
    for (int idx = tid; idx < 32 * 128; idx += 256) {
        int l = idx >> 7, c = idx & 127;
        float v = 0.0f;
        if (l < LAG) {
            if (c < EDIM) {
                v = emb[st * EDIM + c];
            } else {
                int sh = skill[b * TSEQ + t + l];
                int lb = label[b * TSEQ + t + l];
                float lab = (lb == 0) ? -1.0f : ((lb == -1) ? 0.0f : (float)lb);
                v = emb[sh * EDIM + (c - EDIM)] * lab;
            }
        }
        sX[idx] = (_Float16)v;
    }
    // ---- per-edge weights w_l = adj[b,sh,st] * exp(-dtw)
    if (tid < 32) {
        float wv = 0.0f;
        if (tid < LAG) {
            int   sh  = skill[b * TSEQ + t + tid];
            float th  = (float)time_s[b * TSEQ + t + tid];
            float dtw = logf(fabsf(th - ttg) + 1e-6f) * 0.6213349345596119f; // 1/ln(5)
            float cw  = adj[((size_t)b * NSK + sh) * NSK + st];
            wv = cw * expf(-dtw);
        }
        sW[tid] = wv;
    }
    __syncthreads();

    const v16h* pkW0 = (const v16h*)(pk + P_IVW0);
    const v16h* pkW1 = (const v16h*)(pk + P_IVW1);
    const v16h* pkWf = (const v16h*)(pk + P_IVWF);

    // ---- H1 = leaky(X @ W0 + b0), X: 32x128, W0: 128x64
    {
        v8f acc = {0.f, 0.f, 0.f, 0.f, 0.f, 0.f, 0.f, 0.f};
#pragma unroll
        for (int kt = 0; kt < 4; ++kt) {
            v16h a  = load_a_frag(sX + (mt * 16 + laneM) * 128, kt, kg);
            v16h bf = pkW0[(kt * 4 + nt) * 32 + lane];
            acc = __builtin_amdgcn_wmma_f32_16x16x32_f16(false, a, false, bf,
                                                         (short)0, acc, false, false);
        }
        int n = nt * 16 + laneM;
        float bb = b0[n];
#pragma unroll
        for (int v = 0; v < 8; ++v) {
            int m = mt * 16 + v + kg * 8;
            sH[m * 64 + n] = leaky(acc[v] + bb);
        }
    }
    __syncthreads();

    // ---- H2 = H1 + leaky(LN(H1) @ W1 + b1)
    layernorm_to_f16(sH, sX, g1, be1, sMean, sRstd, tid);
    {
        v8f acc = {0.f, 0.f, 0.f, 0.f, 0.f, 0.f, 0.f, 0.f};
#pragma unroll
        for (int kt = 0; kt < 2; ++kt) {
            v16h a  = load_a_frag(sX + (mt * 16 + laneM) * 64, kt, kg);
            v16h bf = pkW1[(kt * 4 + nt) * 32 + lane];
            acc = __builtin_amdgcn_wmma_f32_16x16x32_f16(false, a, false, bf,
                                                         (short)0, acc, false, false);
        }
        int n = nt * 16 + laneM;
        float bb = b1[n];
        float h2[8];
#pragma unroll
        for (int v = 0; v < 8; ++v) {
            int m = mt * 16 + v + kg * 8;
            h2[v] = sH[m * 64 + n] + leaky(acc[v] + bb);
        }
        __syncthreads();
#pragma unroll
        for (int v = 0; v < 8; ++v) {
            int m = mt * 16 + v + kg * 8;
            sH[m * 64 + n] = h2[v];
        }
    }
    __syncthreads();

    // ---- Mout = LN(H2) @ Wf + bf
    layernorm_to_f16(sH, sX, gf, bef, sMean, sRstd, tid);
    {
        v8f acc = {0.f, 0.f, 0.f, 0.f, 0.f, 0.f, 0.f, 0.f};
#pragma unroll
        for (int kt = 0; kt < 2; ++kt) {
            v16h a  = load_a_frag(sX + (mt * 16 + laneM) * 64, kt, kg);
            v16h bf = pkWf[(kt * 4 + nt) * 32 + lane];
            acc = __builtin_amdgcn_wmma_f32_16x16x32_f16(false, a, false, bf,
                                                         (short)0, acc, false, false);
        }
        int n = nt * 16 + laneM;
        float bb = bfv[n];
        float mo[8];
#pragma unroll
        for (int v = 0; v < 8; ++v) mo[v] = acc[v] + bb;
        __syncthreads();
#pragma unroll
        for (int v = 0; v < 8; ++v) {
            int m = mt * 16 + v + kg * 8;
            sH[m * 64 + n] = mo[v];
        }
    }
    __syncthreads();

    // ---- cross[e] = sum_l w_l * Mout[l][e]
    if (tid < 64) {
        float s = 0.0f;
        for (int l = 0; l < LAG; ++l) s += sW[l] * sH[l * 64 + tid];
        cross[(size_t)r * 64 + tid] = s;
    }
}

// -------------------------------------------------------------------------
// Kernel 2: final residual MLP on 32 rows/block of cross_effect -> softmax.
// -------------------------------------------------------------------------
__global__ __launch_bounds__(256)
void final_mlp_kernel(const float* __restrict__ cross,
                      const float* __restrict__ b0, const float* __restrict__ g1,
                      const float* __restrict__ be1, const float* __restrict__ b1,
                      const float* __restrict__ gf, const float* __restrict__ bef,
                      const float* __restrict__ bfin,
                      const _Float16* __restrict__ pk,
                      float* __restrict__ out) {
    __shared__ _Float16 sX[32 * 64];
    __shared__ float    sC[32 * 64];
    __shared__ float    sH[32 * 64];
    __shared__ float    sL[32 * 2];
    __shared__ float    sMean[32], sRstd[32];

    const int r0   = blockIdx.x * 32;
    const int tid  = threadIdx.x;
    const int wave = tid >> 5, lane = tid & 31;
    const int mt = wave >> 2, nt = wave & 3;
    const int laneM = lane & 15, kg = lane >> 4;

    for (int idx = tid; idx < 32 * 64; idx += 256) {
        float v = cross[(size_t)r0 * 64 + idx];
        sC[idx] = v;
        sX[idx] = (_Float16)v;
    }
    __syncthreads();

    const v16h* pkW0 = (const v16h*)(pk + P_FOW0);
    const v16h* pkW1 = (const v16h*)(pk + P_FOW1);
    const v16h* pkWf = (const v16h*)(pk + P_FOWF);

    // ---- H1 = x + leaky(x @ W0 + b0)   (res_first=True)
    {
        v8f acc = {0.f, 0.f, 0.f, 0.f, 0.f, 0.f, 0.f, 0.f};
#pragma unroll
        for (int kt = 0; kt < 2; ++kt) {
            v16h a  = load_a_frag(sX + (mt * 16 + laneM) * 64, kt, kg);
            v16h bf = pkW0[(kt * 4 + nt) * 32 + lane];
            acc = __builtin_amdgcn_wmma_f32_16x16x32_f16(false, a, false, bf,
                                                         (short)0, acc, false, false);
        }
        int n = nt * 16 + laneM;
        float bb = b0[n];
#pragma unroll
        for (int v = 0; v < 8; ++v) {
            int m = mt * 16 + v + kg * 8;
            sH[m * 64 + n] = sC[m * 64 + n] + leaky(acc[v] + bb);
        }
    }
    __syncthreads();

    // ---- H2 = H1 + leaky(LN(H1) @ W1 + b1)
    layernorm_to_f16(sH, sX, g1, be1, sMean, sRstd, tid);
    {
        v8f acc = {0.f, 0.f, 0.f, 0.f, 0.f, 0.f, 0.f, 0.f};
#pragma unroll
        for (int kt = 0; kt < 2; ++kt) {
            v16h a  = load_a_frag(sX + (mt * 16 + laneM) * 64, kt, kg);
            v16h bf = pkW1[(kt * 4 + nt) * 32 + lane];
            acc = __builtin_amdgcn_wmma_f32_16x16x32_f16(false, a, false, bf,
                                                         (short)0, acc, false, false);
        }
        int n = nt * 16 + laneM;
        float bb = b1[n];
        float h2[8];
#pragma unroll
        for (int v = 0; v < 8; ++v) {
            int m = mt * 16 + v + kg * 8;
            h2[v] = sH[m * 64 + n] + leaky(acc[v] + bb);
        }
        __syncthreads();
#pragma unroll
        for (int v = 0; v < 8; ++v) {
            int m = mt * 16 + v + kg * 8;
            sH[m * 64 + n] = h2[v];
        }
    }
    __syncthreads();

    // ---- logits = LN(H2) @ Wf + bf  (64x2, padded to 64x16; only nt==0 waves)
    layernorm_to_f16(sH, sX, gf, bef, sMean, sRstd, tid);
    if (nt == 0) {                       // wave-uniform branch: EXEC all-ones
        v8f acc = {0.f, 0.f, 0.f, 0.f, 0.f, 0.f, 0.f, 0.f};
#pragma unroll
        for (int kt = 0; kt < 2; ++kt) {
            v16h a  = load_a_frag(sX + (mt * 16 + laneM) * 64, kt, kg);
            v16h bf = pkWf[kt * 32 + lane];
            acc = __builtin_amdgcn_wmma_f32_16x16x32_f16(false, a, false, bf,
                                                         (short)0, acc, false, false);
        }
        if (laneM < 2) {
#pragma unroll
            for (int v = 0; v < 8; ++v) {
                int m = mt * 16 + v + kg * 8;
                sL[m * 2 + laneM] = acc[v] + bfin[laneM];
            }
        }
    }
    __syncthreads();

    // ---- softmax over 2 classes, keep class 1
    if (tid < 32) {
        float l0 = sL[tid * 2 + 0], l1 = sL[tid * 2 + 1];
        float mx = fmaxf(l0, l1);
        float e0 = expf(l0 - mx), e1 = expf(l1 - mx);
        out[r0 + tid] = e1 / (e0 + e1);
    }
}

// -------------------------------------------------------------------------
extern "C" void kernel_launch(void* const* d_in, const int* in_sizes, int n_in,
                              void* d_out, int out_size, void* d_ws, size_t ws_size,
                              hipStream_t stream) {
    const int*   skill  = (const int*)  d_in[0];
    // d_in[1] = problem_seq (unused by the reference math)
    const int*   time_s = (const int*)  d_in[2];
    const int*   label  = (const int*)  d_in[3];
    const float* adj    = (const float*)d_in[4];
    const float* emb    = (const float*)d_in[5];
    const float* ivW0 = (const float*)d_in[6];  const float* ivb0  = (const float*)d_in[7];
    const float* ivg1 = (const float*)d_in[8];  const float* ivbe1 = (const float*)d_in[9];
    const float* ivW1 = (const float*)d_in[10]; const float* ivb1  = (const float*)d_in[11];
    const float* ivgf = (const float*)d_in[12]; const float* ivbef = (const float*)d_in[13];
    const float* ivWf = (const float*)d_in[14]; const float* ivbf  = (const float*)d_in[15];
    const float* foW0 = (const float*)d_in[16]; const float* fob0  = (const float*)d_in[17];
    const float* fog1 = (const float*)d_in[18]; const float* fobe1 = (const float*)d_in[19];
    const float* foW1 = (const float*)d_in[20]; const float* fob1  = (const float*)d_in[21];
    const float* fogf = (const float*)d_in[22]; const float* fobef = (const float*)d_in[23];
    const float* foWf = (const float*)d_in[24]; const float* fobf  = (const float*)d_in[25];

    float*    cross = (float*)d_ws;                               // RR x 64 f32
    _Float16* pk    = (_Float16*)((char*)d_ws + CROSS_BYTES);     // packed f16 weights

    pack_weights<<<(P_TOTAL + 255) / 256, 256, 0, stream>>>(
        ivW0, ivW1, ivWf, foW0, foW1, foWf, pk);

    edge_msg_kernel<<<RR, 256, 0, stream>>>(
        skill, time_s, label, adj, emb,
        ivb0, ivg1, ivbe1, ivb1, ivgf, ivbef, ivbf, pk, cross);

    final_mlp_kernel<<<RR / 32, 256, 0, stream>>>(
        cross, fob0, fog1, fobe1, fob1, fogf, fobef, fobf, pk, (float*)d_out);
}
